// MultiHeadSelfAttention_85521388798556
// MI455X (gfx1250) — matrix operand order repaired and checked by reading, not compiled, not run on hardware
//
#include <hip/hip_runtime.h>

// ---------------------------------------------------------------------------
// MI455X (gfx1250) multi-head attention, bf16 WMMA pipeline.
// B=8, N=1024, C=768, H=12, HD=64. ~64 GFLOP vs ~60MB traffic -> compute
// bound, so everything runs through v_wmma_f32_16x16x32_bf16 (f32 accum).
// Working set (~68MB bf16) is L2-resident (192MB): GEMMs use direct global
// loads; attention stages K/V via async global->LDS (ASYNCcnt) double
// buffering, and relays P through LDS for the softmax C->B fragment relayout.
// ---------------------------------------------------------------------------

#define BB 8
#define NN 1024
#define CC 768
#define HH 12
#define HD 64
#define C3 2304            // 3*C
#define MM (BB * NN)       // 8192 rows
#define SCALE 0.125f       // HD^-0.5
#define LDK 72             // padded LDS row stride (shorts): 144B, 8B aligned,
                           // stride 36 dwords -> 16 distinct banks for frag reads

typedef __bf16 v16bf __attribute__((ext_vector_type(16)));
typedef float  v8f   __attribute__((ext_vector_type(8)));
typedef int    v2i   __attribute__((ext_vector_type(2)));

union Frag8 { v8f v; float f[8]; };
union FragB { v16bf v; unsigned int u[8]; unsigned short e[16]; };

#if defined(__AMDGCN__) && __has_builtin(__builtin_amdgcn_global_load_async_to_lds_b64)
#define HAVE_ASYNC_LDS 1
#else
#define HAVE_ASYNC_LDS 0
#endif

__device__ __forceinline__ void wait_async0() {
#if __has_builtin(__builtin_amdgcn_s_wait_asynccnt)
  __builtin_amdgcn_s_wait_asynccnt(0);
#elif defined(__AMDGCN__)
  asm volatile("s_wait_asynccnt 0" ::: "memory");
#endif
}

#if HAVE_ASYNC_LDS
__device__ __forceinline__ void async_b64(const unsigned short* g, unsigned short* l) {
  __builtin_amdgcn_global_load_async_to_lds_b64(
      (__attribute__((address_space(1))) v2i*)g,
      (__attribute__((address_space(3))) v2i*)l, 0, 0);
}
#endif

__device__ __forceinline__ unsigned short f2bf(float x) {
  union { float f; unsigned int u; } c; c.f = x;
  unsigned int r = c.u + 0x7fffu + ((c.u >> 16) & 1u);   // round-nearest-even
  return (unsigned short)(r >> 16);
}

// A-matrix 16x32 bf16 fragment (ISA 7.12.2): lane row = lane%16,
// halves h -> K = ((h<8)?0:16) + 8*(lane/16) + (h&7); pairs are contiguous.
__device__ __forceinline__ v16bf load_frag_a(const unsigned short* p, int g) {
  FragB fr;
#pragma unroll
  for (int r = 0; r < 8; ++r) {
    int k0 = ((r < 4) ? (2 * r) : (2 * r + 8)) + 8 * g;
    fr.u[r] = *(const unsigned int*)(p + k0);
  }
  return fr.v;
}

// B-matrix 32x16 bf16 fragment: lane column = lane%16, same K order as A:
// K = ((h<8)?0:16) + 8*(lane/16) + (h&7).
__device__ __forceinline__ v16bf load_frag_b(const unsigned short* p, int g) {
  FragB fr;
#pragma unroll
  for (int r = 0; r < 8; ++r) {
    int k0 = ((r < 4) ? (2 * r) : (2 * r + 8)) + 8 * g;
    fr.u[r] = *(const unsigned int*)(p + k0);
  }
  return fr.v;
}

__device__ __forceinline__ v8f wmma_bf16(v16bf a, v16bf b, v8f c) {
  // (neg_a, A, neg_b, B, c_mod, C, reuse_a, reuse_b)
  return __builtin_amdgcn_wmma_f32_16x16x32_bf16(false, a, false, b, (short)0, c,
                                                 false, false);
}

// ---------------------------------------------------------------------------
// f32 -> bf16 convert / transposed convert
// ---------------------------------------------------------------------------
__global__ void k_cvt_bf16(const float* __restrict__ in,
                           unsigned short* __restrict__ out, int n) {
  int i = blockIdx.x * 256 + threadIdx.x;
  if (i < n) out[i] = f2bf(in[i]);
}

// in: K x Nc (row-major f32)  ->  out: Nc x K (row-major bf16)
__global__ void k_transpose_bf16(const float* __restrict__ in,
                                 unsigned short* __restrict__ out, int K, int Nc) {
  long long i = (long long)blockIdx.x * 256 + threadIdx.x;
  long long total = (long long)K * Nc;
  if (i >= total) return;
  int n = (int)(i / K), k = (int)(i % K);
  out[i] = f2bf(in[(long long)k * Nc + n]);
}

// ---------------------------------------------------------------------------
// GEMM: C[M,Ncols] = A[M,K](bf16) * Bt[Ncols,K](bf16)^T.
// Block = 256 thr = 8 waves; block tile 256M x 64N; wave tile 32M x 64N
// (2 A-frags x 4 B-frags -> 8 wmma per 32-deep K step).
// ---------------------------------------------------------------------------
__global__ __launch_bounds__(256) void k_gemm_bf16out(
    const unsigned short* __restrict__ A, const unsigned short* __restrict__ Bt,
    unsigned short* __restrict__ Cmat, int K, int Ncols) {
  const int lane = threadIdx.x & 31, wid = threadIdx.x >> 5;
  const int g = lane >> 4, ln = lane & 15;
  const int mrow = blockIdx.x * 256 + wid * 32;
  const int nbase = blockIdx.y * 64;

  Frag8 acc[2][4];
#pragma unroll
  for (int m = 0; m < 2; ++m)
#pragma unroll
    for (int t = 0; t < 4; ++t)
#pragma unroll
      for (int j = 0; j < 8; ++j) acc[m][t].f[j] = 0.0f;

  const unsigned short* arow0 = A + (size_t)(mrow + ln) * K;
  const unsigned short* arow1 = A + (size_t)(mrow + 16 + ln) * K;
  for (int kb = 0; kb < K; kb += 32) {
    v16bf a0 = load_frag_a(arow0 + kb, g);
    v16bf a1 = load_frag_a(arow1 + kb, g);
#pragma unroll
    for (int t = 0; t < 4; ++t) {
      v16bf bf_ = load_frag_b(Bt + (size_t)(nbase + 16 * t + ln) * K + kb, g);
      acc[0][t].v = wmma_bf16(a0, bf_, acc[0][t].v);
      acc[1][t].v = wmma_bf16(a1, bf_, acc[1][t].v);
    }
  }
#pragma unroll
  for (int m = 0; m < 2; ++m)
#pragma unroll
    for (int t = 0; t < 4; ++t)
#pragma unroll
      for (int j = 0; j < 8; ++j)
        Cmat[(size_t)(mrow + 16 * m + j + 8 * g) * Ncols + nbase + 16 * t + ln] =
            f2bf(acc[m][t].f[j]);
}

// Same GEMM but f32 output + bias (final projection).
__global__ __launch_bounds__(256) void k_gemm_f32out(
    const unsigned short* __restrict__ A, const unsigned short* __restrict__ Bt,
    const float* __restrict__ bias, float* __restrict__ Cmat, int K, int Ncols) {
  const int lane = threadIdx.x & 31, wid = threadIdx.x >> 5;
  const int g = lane >> 4, ln = lane & 15;
  const int mrow = blockIdx.x * 256 + wid * 32;
  const int nbase = blockIdx.y * 64;

  Frag8 acc[2][4];
#pragma unroll
  for (int m = 0; m < 2; ++m)
#pragma unroll
    for (int t = 0; t < 4; ++t)
#pragma unroll
      for (int j = 0; j < 8; ++j) acc[m][t].f[j] = 0.0f;

  const unsigned short* arow0 = A + (size_t)(mrow + ln) * K;
  const unsigned short* arow1 = A + (size_t)(mrow + 16 + ln) * K;
  for (int kb = 0; kb < K; kb += 32) {
    v16bf a0 = load_frag_a(arow0 + kb, g);
    v16bf a1 = load_frag_a(arow1 + kb, g);
#pragma unroll
    for (int t = 0; t < 4; ++t) {
      v16bf bf_ = load_frag_b(Bt + (size_t)(nbase + 16 * t + ln) * K + kb, g);
      acc[0][t].v = wmma_bf16(a0, bf_, acc[0][t].v);
      acc[1][t].v = wmma_bf16(a1, bf_, acc[1][t].v);
    }
  }
#pragma unroll
  for (int m = 0; m < 2; ++m)
#pragma unroll
    for (int t = 0; t < 4; ++t) {
      float bv = bias[nbase + 16 * t + ln];
#pragma unroll
      for (int j = 0; j < 8; ++j)
        Cmat[(size_t)(mrow + 16 * m + j + 8 * g) * Ncols + nbase + 16 * t + ln] =
            acc[m][t].f[j] + bv;
    }
}

// ---------------------------------------------------------------------------
// Async double-buffered staging of one 32-key block of K and V (bf16,
// row-padded to LDK shorts). 512 8-byte chunks per matrix / 256 threads.
// ---------------------------------------------------------------------------
__device__ __forceinline__ void stage_kv(const unsigned short* kg,
                                         const unsigned short* vg,
                                         unsigned short* bK, unsigned short* bV,
                                         int tid) {
#pragma unroll
  for (int s = 0; s < 2; ++s) {
    int ch = tid + s * 256;                 // 0..511
    int r = ch >> 4;                        // key row 0..31
    int c = (ch & 15) << 2;                 // col in shorts (8B chunks)
#if HAVE_ASYNC_LDS
    async_b64(kg + (size_t)r * C3 + c, bK + r * LDK + c);
    async_b64(vg + (size_t)r * C3 + c, bV + r * LDK + c);
#else
    *(unsigned long long*)(bK + r * LDK + c) =
        *(const unsigned long long*)(kg + (size_t)r * C3 + c);
    *(unsigned long long*)(bV + r * LDK + c) =
        *(const unsigned long long*)(vg + (size_t)r * C3 + c);
#endif
  }
}

// ---------------------------------------------------------------------------
// Flash-style attention. qkv: [B*N, 3C] bf16 (Q|K|V packed per row).
// Block = 8 waves; each wave owns 16 queries of one (b,h).
// S^T = K*Q^T (lane = query column -> per-lane softmax stats),
// O^T = V^T * P^T accumulated with online rescaling.
// ---------------------------------------------------------------------------
__global__ __launch_bounds__(256) void k_attention(
    const unsigned short* __restrict__ qkv, unsigned short* __restrict__ attn_out) {
  __shared__ __align__(16) unsigned short ldsK[2][32 * LDK];   // [key][d]
  __shared__ __align__(16) unsigned short ldsV[2][32 * LDK];   // [key][d]
  __shared__ __align__(16) unsigned short ldsP[8][16 * 32];    // per-wave P [q][key]

  const int tid = threadIdx.x, lane = tid & 31, wid = tid >> 5;
  const int g = lane >> 4, ln = lane & 15;
  const int b = blockIdx.z, h = blockIdx.y;
  const int qrow = blockIdx.x * 128 + wid * 16 + ln;

  const unsigned short* qptr =
      qkv + ((size_t)b * NN + qrow) * C3 + h * HD;             // Q row (this query)
  const unsigned short* kglob = qkv + (size_t)b * NN * C3 + h * HD + CC;
  const unsigned short* vglob = kglob + CC;

  // Q^T B-fragments (d = 0..31 and 32..63), kept in registers.
  v16bf bq0 = load_frag_b(qptr, g);
  v16bf bq1 = load_frag_b(qptr + 32, g);

  float m_run = -3.0e38f, l_run = 0.0f;
  Frag8 o[4];
#pragma unroll
  for (int t = 0; t < 4; ++t)
#pragma unroll
    for (int j = 0; j < 8; ++j) o[t].f[j] = 0.0f;

  const int NIT = NN / 32;
  stage_kv(kglob, vglob, ldsK[0], ldsV[0], tid);   // preload tile 0

  for (int it = 0; it < NIT; ++it) {
    const int cur = it & 1;
    wait_async0();          // own async chunks for tile `it` have landed
    __syncthreads();        // everyone's chunks landed; prev buffer free
    if (it + 1 < NIT)
      stage_kv(kglob + (size_t)(it + 1) * 32 * C3,
               vglob + (size_t)(it + 1) * 32 * C3,
               ldsK[1 - cur], ldsV[1 - cur], tid);   // overlap with compute

    const unsigned short* bKc = ldsK[cur];
    const unsigned short* bVc = ldsV[cur];

    // S^T tiles: keys [16t .. 16t+15] x 16 queries.
    Frag8 st[2];
#pragma unroll
    for (int t = 0; t < 2; ++t) {
      v16bf ak0 = load_frag_a(bKc + (16 * t + ln) * LDK, g);
      v16bf ak1 = load_frag_a(bKc + (16 * t + ln) * LDK + 32, g);
      Frag8 z;
#pragma unroll
      for (int j = 0; j < 8; ++j) z.f[j] = 0.0f;
      z.v = wmma_bf16(ak0, bq0, z.v);
      z.v = wmma_bf16(ak1, bq1, z.v);
      st[t] = z;
    }

    // Online softmax over this 32-key block (lane owns query q = ln).
    float sv[16];
    float vmax = -3.0e38f;
#pragma unroll
    for (int t = 0; t < 2; ++t)
#pragma unroll
      for (int j = 0; j < 8; ++j) {
        float s = st[t].f[j] * SCALE;
        sv[t * 8 + j] = s;
        vmax = fmaxf(vmax, s);
      }
    vmax = fmaxf(vmax, __shfl_xor(vmax, 16, 32));   // combine lane pair (same q)
    float m_new = fmaxf(m_run, vmax);
    float alpha = __expf(m_run - m_new);
    float lsum = 0.0f;
#pragma unroll
    for (int i = 0; i < 16; ++i) {
      float p = __expf(sv[i] - m_new);
      sv[i] = p;
      lsum += p;
    }
    lsum += __shfl_xor(lsum, 16, 32);
    l_run = l_run * alpha + lsum;
    m_run = m_new;
#pragma unroll
    for (int t = 0; t < 4; ++t)
#pragma unroll
      for (int j = 0; j < 8; ++j) o[t].f[j] *= alpha;

    // Relayout P through per-wave LDS: store P[q][key], re-read as B frag.
    unsigned short* P = ldsP[wid];
#pragma unroll
    for (int t = 0; t < 2; ++t)
#pragma unroll
      for (int j = 0; j < 8; ++j)
        P[ln * 32 + 16 * t + j + 8 * g] = f2bf(sv[t * 8 + j]);
    v16bf pb = load_frag_b(P + ln * 32, g);   // same-wave LDS ops are in order

    // O^T[dtile] += V^T(16d x 32key) * P^T(32key x 16q).
    // V^T A-frag gathered from untransposed ldsV: elem(d, k) = V[k][d].
#pragma unroll
    for (int dt = 0; dt < 4; ++dt) {
      FragB fv;
      const int d = 16 * dt + ln;
#pragma unroll
      for (int hh = 0; hh < 16; ++hh) {
        int k = ((hh < 8) ? 0 : 16) + 8 * g + (hh & 7);
        fv.e[hh] = bVc[k * LDK + d];
      }
      o[dt].v = wmma_bf16(fv.v, pb, o[dt].v);
    }
    __syncthreads();        // done reading `cur` before it is refilled at it+2
  }

  // Epilogue: normalize and store [b, q, h*64 + d] as bf16.
  float inv = 1.0f / l_run;
  unsigned short* orow = attn_out + ((size_t)b * NN + qrow) * CC + h * HD;
#pragma unroll
  for (int dt = 0; dt < 4; ++dt)
#pragma unroll
    for (int j = 0; j < 8; ++j)
      orow[16 * dt + j + 8 * g] = f2bf(o[dt].f[j] * inv);
}

// ---------------------------------------------------------------------------
extern "C" void kernel_launch(void* const* d_in, const int* in_sizes, int n_in,
                              void* d_out, int out_size, void* d_ws, size_t ws_size,
                              hipStream_t stream) {
  const float* x     = (const float*)d_in[0];   // [8,1024,768]
  const float* w_qkv = (const float*)d_in[1];   // [768,2304]
  const float* w_out = (const float*)d_in[2];   // [768,768]
  const float* b_out = (const float*)d_in[3];   // [768]
  float* out = (float*)d_out;                   // [8,1024,768] f32

  char* ws = (char*)d_ws;
  size_t off = 0;
  auto carve = [&](size_t bytes) {
    void* p = ws + off;
    off = (off + bytes + 255) & ~(size_t)255;
    return p;
  };
  unsigned short* xb     = (unsigned short*)carve((size_t)MM * CC * 2); // 12.6MB
  unsigned short* wqkvT  = (unsigned short*)carve((size_t)C3 * CC * 2); //  3.5MB
  unsigned short* woutT  = (unsigned short*)carve((size_t)CC * CC * 2); //  1.2MB
  unsigned short* qkvb   = (unsigned short*)carve((size_t)MM * C3 * 2); // 37.7MB
  unsigned short* attnb  = (unsigned short*)carve((size_t)MM * CC * 2); // 12.6MB

  // 1) precision conversion / weight transposes
  {
    int n = MM * CC;
    k_cvt_bf16<<<(n + 255) / 256, 256, 0, stream>>>(x, xb, n);
  }
  {
    long long n = (long long)CC * C3;
    k_transpose_bf16<<<(unsigned)((n + 255) / 256), 256, 0, stream>>>(w_qkv, wqkvT, CC, C3);
  }
  {
    long long n = (long long)CC * CC;
    k_transpose_bf16<<<(unsigned)((n + 255) / 256), 256, 0, stream>>>(w_out, woutT, CC, CC);
  }

  // 2) QKV projection: [8192,768] x [768,2304] -> bf16 [8192,2304]
  k_gemm_bf16out<<<dim3(MM / 256, C3 / 64), 256, 0, stream>>>(xb, wqkvT, qkvb, CC, C3);

  // 3) fused attention (flash-style, online softmax, async K/V staging)
  k_attention<<<dim3(NN / 128, HH, BB), 256, 0, stream>>>(qkvb, attnb);

  // 4) output projection + bias: [8192,768] x [768,768] -> f32
  k_gemm_f32out<<<dim3(MM / 256, CC / 64), 256, 0, stream>>>(attnb, woutT, b_out, out, CC, CC);
}
